// Step_4423816315424
// MI455X (gfx1250) — compile-verified, matrix operand order checked by reading, not run on hardware
//
#include <hip/hip_runtime.h>
#include <hip/hip_bf16.h>

// ---------------------------------------------------------------------------
// Types for CDNA5 WMMA (wave32): bf16 A/B fragments, f32 accumulator.
// ---------------------------------------------------------------------------
typedef __attribute__((ext_vector_type(16))) __bf16 v16bf;
typedef __attribute__((ext_vector_type(8)))  __bf16 v8bf;
typedef __attribute__((ext_vector_type(8)))  float  v8f;

#define DIMC 256
#define NNODES 50000
#define NEDGES 200000
#define NTREE  80000

__device__ __forceinline__ __bf16 f32_to_bf16(float f) {
    unsigned u = __builtin_bit_cast(unsigned, f);
    unsigned r = (u + 0x7FFFu + ((u >> 16) & 1u)) >> 16;
    unsigned short s = (unsigned short)r;
    return __builtin_bit_cast(__bf16, s);
}
__device__ __forceinline__ float bf16_to_f32(__bf16 h) {
    unsigned short s = __builtin_bit_cast(unsigned short, h);
    unsigned u = ((unsigned)s) << 16;
    return __builtin_bit_cast(float, u);
}
__device__ __forceinline__ v16bf load_frag(const __bf16* p) {
    // elements 0..7  = K[kbase .. kbase+7]
    // elements 8..15 = K[kbase+16 .. kbase+23]   (ISA 16-bit A-matrix layout)
    v8bf lo = *(const v8bf*)(p);
    v8bf hi = *(const v8bf*)(p + 16);
    return __builtin_shufflevector(lo, hi, 0,1,2,3,4,5,6,7,8,9,10,11,12,13,14,15);
}

// ---------------------------------------------------------------------------
// WMMA GEMM: Out[M,256](bf16) = A[M,256](bf16) @ W[256,256](bf16)^T + bias
// W is row-major [n,k] exactly like the reference's fc weights, so loading
// W rows with the A-fragment lane pattern supplies B = W^T to the WMMA.
// Block: 8 waves -> 64(M) x 256(N) tile. Wave (wm,wn): 32 x 64 via 2x4 accs.
// Tail handling: M is always a multiple of 16; an out-of-range 16-row subtile
// has its A pointer clamped to row 0 (loads always valid, results discarded),
// keeping the K-loop branch-free so WMMAs pipeline cleanly.
// ---------------------------------------------------------------------------
__global__ __launch_bounds__(256)
void wmma_gemm_bf16(const __bf16* __restrict__ A,
                    const __bf16* __restrict__ W,
                    const float*  __restrict__ bias,
                    __bf16* __restrict__ Out,
                    int M)
{
    constexpr int K = DIMC;
    const int tid  = threadIdx.x;
    const int wave = tid >> 5;
    const int lane = tid & 31;
    const int wm   = wave >> 2;      // 0..1
    const int wn   = wave & 3;       // 0..3
    const long m0  = (long)blockIdx.x * 64 + (long)wm * 32;
    const int  n0  = wn * 64;
    if (m0 >= (long)M) return;                     // uniform within wave
    const bool mv1 = (m0 + 32) <= (long)M;         // second 16-row subtile valid

    const int rsel  = lane & 15;                   // row within 16-tile
    const int kbase = (lane >> 4) << 3;            // 0 or 8

    v8f acc[2][4];
#pragma unroll
    for (int mi = 0; mi < 2; ++mi)
#pragma unroll
        for (int ni = 0; ni < 4; ++ni)
            acc[mi][ni] = (v8f){0.f,0.f,0.f,0.f,0.f,0.f,0.f,0.f};

    const __bf16* arow0 = A + (m0 + rsel) * K + kbase;
    // Clamp invalid subtile to row 0: always-legal loads, branch-free loop.
    const long m1row = mv1 ? (m0 + 16 + rsel) : (long)rsel;
    const __bf16* arow1 = A + m1row * K + kbase;

#pragma unroll
    for (int k0 = 0; k0 < K; k0 += 32) {
        if (k0 + 32 < K) __builtin_prefetch(arow0 + k0 + 32, 0, 3);

        v16bf af[2];
        af[0] = load_frag(arow0 + k0);
        af[1] = load_frag(arow1 + k0);

        v16bf bfr[4];
#pragma unroll
        for (int ni = 0; ni < 4; ++ni)
            bfr[ni] = load_frag(W + (long)(n0 + ni * 16 + rsel) * K + k0 + kbase);

#pragma unroll
        for (int mi = 0; mi < 2; ++mi)
#pragma unroll
            for (int ni = 0; ni < 4; ++ni)
                acc[mi][ni] = __builtin_amdgcn_wmma_f32_16x16x32_bf16(
                    false, af[mi], false, bfr[ni], (short)0, acc[mi][ni],
                    false, false);
    }

    // C/D layout: VGPR v -> M = v + 8*(lane>>4); N = lane&15
    const int rowoff = (lane >> 4) << 3;
    const int col    = lane & 15;
#pragma unroll
    for (int mi = 0; mi < 2; ++mi) {
        if (mi == 1 && !mv1) continue;             // discard clamped subtile
#pragma unroll
        for (int ni = 0; ni < 4; ++ni) {
            const int c  = n0 + ni * 16 + col;
            const float bv = bias ? bias[c] : 0.0f;
#pragma unroll
            for (int v = 0; v < 8; ++v) {
                const long r = m0 + mi * 16 + rowoff + v;
                Out[r * DIMC + c] = f32_to_bf16(acc[mi][ni][v] + bv);
            }
        }
    }
}

// ---------------------------------------------------------------------------
// Elementwise / gather / reduction helpers
// ---------------------------------------------------------------------------
__global__ void cast_f32_bf16_kernel(const float* __restrict__ in,
                                     __bf16* __restrict__ out, long n8)
{
    long g = (long)blockIdx.x * blockDim.x + threadIdx.x;
    if (g >= n8) return;
    const float4* p = (const float4*)(in + g * 8);
    float4 a = p[0], b = p[1];
    v8bf o;
    o[0] = f32_to_bf16(a.x); o[1] = f32_to_bf16(a.y);
    o[2] = f32_to_bf16(a.z); o[3] = f32_to_bf16(a.w);
    o[4] = f32_to_bf16(b.x); o[5] = f32_to_bf16(b.y);
    o[6] = f32_to_bf16(b.z); o[7] = f32_to_bf16(b.w);
    *(v8bf*)(out + g * 8) = o;
}

__global__ void copy_f32_kernel(const float* __restrict__ in,
                                float* __restrict__ out, long n4)
{
    long g = (long)blockIdx.x * blockDim.x + threadIdx.x;
    if (g >= n4) return;
    ((float4*)out)[g] = ((const float4*)in)[g];
}

// h[row] = P0[idx[row*stride+0]] + P1[idx[row*stride+1]]   (bf16, 8 feat/thread)
__global__ void gather2_kernel(const __bf16* __restrict__ P0,
                               const __bf16* __restrict__ P1,
                               const int* __restrict__ idx,
                               __bf16* __restrict__ out, long rows)
{
    long g = (long)blockIdx.x * blockDim.x + threadIdx.x;
    if (g >= rows * 32) return;
    long row = g >> 5;
    int  f8  = (int)(g & 31) << 3;
    long i0 = idx[row * 2 + 0], i1 = idx[row * 2 + 1];
    v8bf a = *(const v8bf*)(P0 + i0 * DIMC + f8);
    v8bf b = *(const v8bf*)(P1 + i1 * DIMC + f8);
    v8bf o;
#pragma unroll
    for (int j = 0; j < 8; ++j)
        o[j] = f32_to_bf16(bf16_to_f32(a[j]) + bf16_to_f32(b[j]));
    *(v8bf*)(out + row * DIMC + f8) = o;
}

__global__ void gather3_kernel(const __bf16* __restrict__ P0,
                               const __bf16* __restrict__ P1,
                               const __bf16* __restrict__ P2,
                               const int* __restrict__ idx,
                               __bf16* __restrict__ out, long rows)
{
    long g = (long)blockIdx.x * blockDim.x + threadIdx.x;
    if (g >= rows * 32) return;
    long row = g >> 5;
    int  f8  = (int)(g & 31) << 3;
    long i0 = idx[row * 3 + 0], i1 = idx[row * 3 + 1], i2 = idx[row * 3 + 2];
    v8bf a = *(const v8bf*)(P0 + i0 * DIMC + f8);
    v8bf b = *(const v8bf*)(P1 + i1 * DIMC + f8);
    v8bf c = *(const v8bf*)(P2 + i2 * DIMC + f8);
    v8bf o;
#pragma unroll
    for (int j = 0; j < 8; ++j)
        o[j] = f32_to_bf16(bf16_to_f32(a[j]) + bf16_to_f32(b[j]) + bf16_to_f32(c[j]));
    *(v8bf*)(out + row * DIMC + f8) = o;
}

__global__ void zero_stats_kernel(float* sum, float* sq)
{
    int t = threadIdx.x;
    sum[t] = 0.0f; sq[t] = 0.0f;
}

// per-feature sum & sumsq (feature = threadIdx.x), grid-stride over rows
__global__ void stats_kernel(const __bf16* __restrict__ X, long M,
                             float* __restrict__ sum, float* __restrict__ sq)
{
    int f = threadIdx.x;
    float s = 0.0f, q = 0.0f;
    for (long r = blockIdx.x; r < M; r += gridDim.x) {
        float v = bf16_to_f32(X[r * DIMC + f]);
        s += v; q += v * v;
    }
    atomicAdd(&sum[f], s);
    atomicAdd(&sq[f],  q);
}

__global__ void finalize_stats_kernel(const float* __restrict__ sum,
                                      const float* __restrict__ sq, long M,
                                      float* __restrict__ mean,
                                      float* __restrict__ rstd)
{
    int t = threadIdx.x;
    float m = sum[t] / (float)M;
    float v = sq[t] / (float)M - m * m;
    mean[t] = m;
    rstd[t] = rsqrtf(v + 1e-5f);
}

template <bool F32OUT>
__global__ void bn_relu_kernel(const __bf16* __restrict__ X,
                               const float* __restrict__ mean,
                               const float* __restrict__ rstd,
                               const float* __restrict__ gam,
                               const float* __restrict__ bet,
                               void* __restrict__ out, long rows)
{
    long g = (long)blockIdx.x * blockDim.x + threadIdx.x;
    if (g >= rows * 32) return;
    long row = g >> 5;
    int  f8  = (int)(g & 31) << 3;
    v8bf x = *(const v8bf*)(X + row * DIMC + f8);
    float y[8];
#pragma unroll
    for (int j = 0; j < 8; ++j) {
        int f = f8 + j;
        float v = (bf16_to_f32(x[j]) - mean[f]) * rstd[f] * gam[f] + bet[f];
        y[j] = v > 0.0f ? v : 0.0f;
    }
    if (F32OUT) {
        float4* o = (float4*)((float*)out + row * DIMC + f8);
        o[0] = make_float4(y[0], y[1], y[2], y[3]);
        o[1] = make_float4(y[4], y[5], y[6], y[7]);
    } else {
        v8bf o;
#pragma unroll
        for (int j = 0; j < 8; ++j) o[j] = f32_to_bf16(y[j]);
        *(v8bf*)((__bf16*)out + row * DIMC + f8) = o;
    }
}

// acc[idx[row*stride+comp]] += H[row]  (f32 atomics = segment_sum)
__global__ void scatter_add_kernel(const __bf16* __restrict__ H,
                                   const int* __restrict__ idx, int stride, int comp,
                                   float* __restrict__ acc, long rows)
{
    long g = (long)blockIdx.x * blockDim.x + threadIdx.x;
    if (g >= rows * 32) return;
    long row = g >> 5;
    int  f8  = (int)(g & 31) << 3;
    long node = idx[row * stride + comp];
    v8bf h = *(const v8bf*)(H + row * DIMC + f8);
    float* dst = acc + node * DIMC + f8;
#pragma unroll
    for (int j = 0; j < 8; ++j)
        atomicAdd(&dst[j], bf16_to_f32(h[j]));
}

// ---------------------------------------------------------------------------
// Host orchestration
// ---------------------------------------------------------------------------
static inline unsigned gblk(long total, int bs) { return (unsigned)((total + bs - 1) / bs); }

extern "C" void kernel_launch(void* const* d_in, const int* in_sizes, int n_in,
                              void* d_out, int out_size, void* d_ws, size_t ws_size,
                              hipStream_t stream)
{
    (void)in_sizes; (void)n_in; (void)out_size; (void)ws_size;

    const float* x        = (const float*)d_in[0];
    const int*   edges    = (const int*)d_in[1];
    const int*   treelets = (const int*)d_in[2];
    auto P = [&](int i) -> const float* { return (const float*)d_in[i]; };

    const long N = NNODES, E = NEDGES, T = NTREE;
    const long NODE_BF = N * DIMC * 2;   // bytes
    const long NODE_F  = N * DIMC * 4;
    const long EDGE_BF = E * DIMC * 2;

    // ---- workspace carve (256B aligned sizes) ----
    char* w = (char*)d_ws;
    __bf16* xb    = (__bf16*)w;              w += NODE_BF;            // x in bf16
    float*  acc   = (float*)w;               w += NODE_F;             // x_new accumulator
    __bf16* wslot = (__bf16*)w;              w += DIMC * DIMC * 2;    // current weight, bf16
    __bf16* p0    = (__bf16*)w;              w += NODE_BF;
    __bf16* p1    = (__bf16*)w;              w += NODE_BF;
    __bf16* p2    = (__bf16*)w;              w += NODE_BF;
    __bf16* hbuf  = (__bf16*)w;              w += EDGE_BF;            // >= T rows too
    __bf16* h2buf = (__bf16*)w;              w += EDGE_BF;
    __bf16* xnb   = (__bf16*)w;              w += NODE_BF;            // x_new bf16
    float*  s_sum = (float*)w;               w += DIMC * 4;
    float*  s_sq  = (float*)w;               w += DIMC * 4;
    float*  s_mu  = (float*)w;               w += DIMC * 4;
    float*  s_rs  = (float*)w;               w += DIMC * 4;

    const int BS = 256;

    auto gemm = [&](const __bf16* A, int widx, const float* bias, __bf16* Out, long M) {
        // cast this GEMM's f32 weight to bf16 (stream order serializes reuse of wslot)
        cast_f32_bf16_kernel<<<gblk((DIMC * DIMC) / 8, BS), BS, 0, stream>>>(
            P(widx), wslot, (DIMC * DIMC) / 8);
        wmma_gemm_bf16<<<gblk(M, 64), BS, 0, stream>>>(A, wslot, bias, Out, (int)M);
    };
    auto bnstats = [&](const __bf16* X, long M) {
        zero_stats_kernel<<<1, DIMC, 0, stream>>>(s_sum, s_sq);
        stats_kernel<<<1024, DIMC, 0, stream>>>(X, M, s_sum, s_sq);
        finalize_stats_kernel<<<1, DIMC, 0, stream>>>(s_sum, s_sq, M, s_mu, s_rs);
    };

    // ---- init: xb = bf16(x); acc = x ----
    cast_f32_bf16_kernel<<<gblk(N * DIMC / 8, BS), BS, 0, stream>>>(x, xb, N * DIMC / 8);
    copy_f32_kernel<<<gblk(N * DIMC / 4, BS), BS, 0, stream>>>(x, acc, N * DIMC / 4);

    // ---- edge stages: FI (agg by e0), FO (agg by e1) ----
    // param idx: fc1a, fc1b, fc1b_b, fc2, fc2_b, bn1_g, bn1_b, bn2_g, bn2_b
    const int ec[2][10] = {
        { 3,  4,  5,  6,  7,  8,  9, 10, 11, 0},
        {12, 13, 14, 15, 16, 17, 18, 19, 20, 1},
    };
    for (int s = 0; s < 2; ++s) {
        gemm(xb, ec[s][0], nullptr,      p0, N);
        gemm(xb, ec[s][1], P(ec[s][2]),  p1, N);
        gather2_kernel<<<gblk(E * 32, BS), BS, 0, stream>>>(p0, p1, edges, hbuf, E);
        bnstats(hbuf, E);
        bn_relu_kernel<false><<<gblk(E * 32, BS), BS, 0, stream>>>(
            hbuf, s_mu, s_rs, P(ec[s][5]), P(ec[s][6]), h2buf, E);
        gemm(h2buf, ec[s][3], P(ec[s][4]), hbuf, E);
        bnstats(hbuf, E);
        bn_relu_kernel<false><<<gblk(E * 32, BS), BS, 0, stream>>>(
            hbuf, s_mu, s_rs, P(ec[s][7]), P(ec[s][8]), h2buf, E);
        scatter_add_kernel<<<gblk(E * 32, BS), BS, 0, stream>>>(
            h2buf, edges, 2, ec[s][9], acc, E);
    }

    // ---- treelet stages: FL (agg t0), FH (agg t1), FR (agg t2) ----
    // param idx: fc1a, fc1b, fc1c, fc1c_b, fc2, fc2_b, bn1_g, bn1_b, bn2_g, bn2_b
    const int tc[3][11] = {
        {21, 22, 23, 24, 25, 26, 27, 28, 29, 30, 0},
        {31, 32, 33, 34, 35, 36, 37, 38, 39, 40, 1},
        {41, 42, 43, 44, 45, 46, 47, 48, 49, 50, 2},
    };
    for (int s = 0; s < 3; ++s) {
        gemm(xb, tc[s][0], nullptr,     p0, N);
        gemm(xb, tc[s][1], nullptr,     p1, N);
        gemm(xb, tc[s][2], P(tc[s][3]), p2, N);
        gather3_kernel<<<gblk(T * 32, BS), BS, 0, stream>>>(p0, p1, p2, treelets, hbuf, T);
        bnstats(hbuf, T);
        bn_relu_kernel<false><<<gblk(T * 32, BS), BS, 0, stream>>>(
            hbuf, s_mu, s_rs, P(tc[s][6]), P(tc[s][7]), h2buf, T);
        gemm(h2buf, tc[s][4], P(tc[s][5]), hbuf, T);
        bnstats(hbuf, T);
        bn_relu_kernel<false><<<gblk(T * 32, BS), BS, 0, stream>>>(
            hbuf, s_mu, s_rs, P(tc[s][8]), P(tc[s][9]), h2buf, T);
        scatter_add_kernel<<<gblk(T * 32, BS), BS, 0, stream>>>(
            h2buf, treelets, 3, tc[s][10], acc, T);
    }

    // ---- final: out = relu(BN(x_new @ FP_w^T + FP_b)) ----
    cast_f32_bf16_kernel<<<gblk(N * DIMC / 8, BS), BS, 0, stream>>>(acc, xnb, N * DIMC / 8);
    gemm(xnb, 51, P(52), hbuf, N);
    bnstats(hbuf, N);
    bn_relu_kernel<true><<<gblk(N * 32, BS), BS, 0, stream>>>(
        hbuf, s_mu, s_rs, P(53), P(54), d_out, N);
}
